// RNNBase_37795712204973
// MI455X (gfx1250) — matrix operand order repaired and checked by reading, not compiled, not run on hardware
//
#include <hip/hip_runtime.h>

#define DEV __device__ __forceinline__

typedef __bf16 bf16_t;
typedef __attribute__((ext_vector_type(8)))  __bf16 bf16x8;
typedef __attribute__((ext_vector_type(16))) __bf16 bf16x16;
typedef __attribute__((ext_vector_type(8)))  float  f32x8;

constexpr int B_ = 64, T_ = 168, M_ = 256, F_ = 32, H_ = 128, O_ = 32;
constexpr int R_ = B_ * M_;      // 16384 independent LSTM rows
constexpr int G_ = 4 * H_;       // 512 gate columns (i,f,g,o)
constexpr float TWO_OVER_PI = 0.63661977236758134f;

// ---------------------------------------------------------------------------
// D = A(16x32 bf16) x B(32x16 bf16) + C(16x16 f32)  -> v_wmma_f32_16x16x32_bf16
DEV f32x8 wmma_bf16(bf16x16 a, bf16x16 b, f32x8 c) {
  return __builtin_amdgcn_wmma_f32_16x16x32_bf16(false, a, false, b,
                                                 (short)0, c, false, false);
}

// Assemble a 16-element bf16 fragment from two contiguous 8-element (16B) runs.
DEV bf16x16 ldpair(const bf16_t* p0, const bf16_t* p1) {
  bf16x8 lo = *(const bf16x8*)p0;
  bf16x8 hi = *(const bf16x8*)p1;
  return __builtin_shufflevector(lo, hi, 0,1,2,3,4,5,6,7,8,9,10,11,12,13,14,15);
}

DEV float sigm(float x) { return 1.0f / (1.0f + __expf(-x)); }

// CDNA5 async global->LDS copy (8B per lane), tracked by ASYNCcnt.
DEV void async_copy_b64(const bf16_t* gsrc, bf16_t* ldst) {
  uint32_t l = (uint32_t)(uintptr_t)ldst;      // LDS byte offset (addr[31:0])
  uint64_t g = (uint64_t)(uintptr_t)gsrc;
  asm volatile("global_load_async_to_lds_b64 %0, %1, off"
               :: "v"(l), "v"(g) : "memory");
}

// ---------------------------------------------------------------------------
// One-time f32 -> bf16 repack of ALL time-invariant weights (incl. assoc_W).
__global__ void pack_k(const float* __restrict__ aW, const float* __restrict__ wih,
                       const float* __restrict__ whh, const float* __restrict__ dw,
                       bf16_t* __restrict__ aW_p, bf16_t* __restrict__ wih_p,
                       bf16_t* __restrict__ whh_p, bf16_t* __restrict__ dw_p) {
  int i = blockIdx.x * blockDim.x + threadIdx.x;
  if (i < F_ * M_ * M_) aW_p[i] = (bf16_t)aW[i];
  if (i < G_ * F_)      wih_p[i] = (bf16_t)wih[i];
  if (i < G_ * H_)      whh_p[i] = (bf16_t)whh[i];
  if (i < O_ * H_)      dw_p[i]  = (bf16_t)dw[i];
}

// ---------------------------------------------------------------------------
// Squash pass: u[a][col][n] = bf16( atan(x[b, t, n, a]) * 2/pi ),
// col = tt*64 + b within the chunk. One block per col; x slice is 32KB
// contiguous (coalesced reads), LDS transpose (padded rows, conflict-free),
// coalesced 512B-per-attribute writes. Each atan computed exactly once.
__global__ __launch_bounds__(256) void squash_k(const float* __restrict__ x,
                                                bf16_t* __restrict__ u_ws,
                                                int chunk, int Tc) {
  __shared__ bf16_t lu[M_ * 34];               // [n][a] padded to 34
  const int col = blockIdx.x;
  const int tt = col >> 6, bb = col & 63;
  const int t = chunk * Tc + tt;
  const float* xp = x + (((size_t)bb * T_ + t) * M_) * F_;   // 8192 floats
  for (int i = threadIdx.x; i < M_ * F_; i += 256) {
    const int n = i >> 5, a = i & 31;
    lu[n * 34 + a] = (bf16_t)(atanf(xp[i]) * TWO_OVER_PI);
  }
  __syncthreads();
  const int cols = Tc * 64;
  #pragma unroll 4
  for (int a = 0; a < F_; ++a)
    u_ws[((size_t)a * cols + col) * M_ + threadIdx.x] = lu[threadIdx.x * 34 + a];
}

// ---------------------------------------------------------------------------
// ICFA GEMM: y[tt, b*M+m, a] = sum_n assoc_W[a,m,n] * u[a][col][n].
// One wave per (a, m-tile, 4 col-tiles): every A fragment is reused for
// 4 WMMAs; all operands are pre-packed bf16 (two 16B loads per fragment).
__global__ __launch_bounds__(128) void icfa_k(const bf16_t* __restrict__ aW_p,
                                              const bf16_t* __restrict__ u_ws,
                                              bf16_t* __restrict__ y_ws,
                                              int Tc) {
  const int lane = threadIdx.x & 31;
  const int wave = threadIdx.x >> 5;
  const int ln = lane & 15;
  const int kh = lane >> 4;
  int tile = blockIdx.x * 4 + wave;            // 512*Tc tiles total
  const int cq = tile % Tc; tile /= Tc;        // quad of col-tiles (one tt)
  const int mt = tile & 15;
  const int a  = tile >> 4;
  const int cols = Tc * 64;

  f32x8 acc[4];
  #pragma unroll
  for (int c4 = 0; c4 < 4; ++c4)
    #pragma unroll
    for (int r = 0; r < 8; ++r) acc[c4][r] = 0.0f;

  const bf16_t* arow = aW_p + ((size_t)a * M_ + mt * 16 + ln) * M_;
  const bf16_t* ubase = u_ws + ((size_t)a * cols + cq * 64) * M_;
  __builtin_prefetch(arow + kh * 128, 0, 1);   // global_prefetch_b8

  #pragma unroll
  for (int kb = 0; kb < 8; ++kb) {
    const bf16_t* ap = arow + kb * 32 + kh * 8;
    const bf16x16 af = ldpair(ap, ap + 16);    // A reused 4x below
    #pragma unroll
    for (int c4 = 0; c4 < 4; ++c4) {
      const bf16_t* up = ubase + (size_t)(c4 * 16 + ln) * M_ + kb * 32 + kh * 8;
      acc[c4] = wmma_bf16(af, ldpair(up, up + 16), acc[c4]);
    }
  }

  #pragma unroll
  for (int c4 = 0; c4 < 4; ++c4) {
    const int bb = c4 * 16 + ln;               // tt == cq for this wave
    #pragma unroll
    for (int r = 0; r < 8; ++r) {
      const int mr = mt * 16 + kh * 8 + r;
      y_ws[((size_t)cq * R_ + (size_t)bb * M_ + mr) * F_ + a] = (bf16_t)acc[c4][r];
    }
  }
}

// ---------------------------------------------------------------------------
// LSTM scan over one t-chunk. Block = 4 waves, owns 16 rows (one b, 16 m's).
// Wave w owns hidden slice [32w,32w+32): i/f/g/o tiles line up lane-wise, so
// the cell update is lane-local. h_t lives in LDS (bf16 A-fragments). The
// 1KB Y tile for step t+1 is double-buffered into LDS with async copies.
__global__ __launch_bounds__(128) void lstm_scan_k(
    const bf16_t* __restrict__ y_ws,
    const bf16_t* __restrict__ wih_p, const bf16_t* __restrict__ whh_p,
    const bf16_t* __restrict__ dw_p,
    const float* __restrict__ b_ih, const float* __restrict__ b_hh,
    const float* __restrict__ db,
    float* __restrict__ out, float* __restrict__ h_ws, float* __restrict__ c_ws,
    int chunk, int Tc) {
  constexpr int HP = H_ + 8;                   // padded LDS row (bank spread)
  __shared__ __align__(16) bf16_t h_s[16 * HP];
  __shared__ __align__(16) bf16_t y_l[2 * 16 * F_];   // double-buffered Y tile

  const int lane = threadIdx.x & 31;
  const int wave = threadIdx.x >> 5;
  const int ln = lane & 15;
  const int kh = lane >> 4;
  const int r0 = blockIdx.x * 16;              // first row (= b*256 + m0)
  const int bb = r0 >> 8;
  const int m0 = r0 & 255;

  // Per-lane gate biases (time-invariant).
  float bias[4][2];
  #pragma unroll
  for (int q = 0; q < 4; ++q)
    #pragma unroll
    for (int u = 0; u < 2; ++u) {
      const int col = q * H_ + wave * 32 + u * 16 + ln;
      bias[q][u] = b_ih[col] + b_hh[col];
    }
  float dbias = 0.0f; int ocol = 0;
  if (wave < 2) { ocol = wave * 16 + ln; dbias = db[ocol]; }

  // Carried cell state (D-fragment layout); h_{chunk-1} restored into LDS.
  f32x8 cfr[2];
  if (chunk == 0) {
    #pragma unroll
    for (int u = 0; u < 2; ++u)
      #pragma unroll
      for (int r = 0; r < 8; ++r) cfr[u][r] = 0.0f;
    for (int i = threadIdx.x; i < 16 * HP; i += 128) h_s[i] = (bf16_t)0.0f;
  } else {
    #pragma unroll
    for (int u = 0; u < 2; ++u)
      #pragma unroll
      for (int r = 0; r < 8; ++r)
        cfr[u][r] = c_ws[(size_t)(r0 + kh * 8 + r) * H_ + wave * 32 + u * 16 + ln];
    for (int i = threadIdx.x; i < 16 * H_; i += 128) {
      const int row = i >> 7, cc = i & 127;
      h_s[row * HP + cc] = (bf16_t)h_ws[(size_t)(r0 + row) * H_ + cc];
    }
  }

  // Kick off async copy of the first Y tile (8B per lane, 1KB per tile).
  const bf16_t* ytile0 = y_ws + (size_t)r0 * F_;
  async_copy_b64(ytile0 + threadIdx.x * 4, &y_l[threadIdx.x * 4]);
  __syncthreads();

  for (int tt = 0; tt < Tc; ++tt) {
    const int buf = tt & 1;
    if (tt + 1 < Tc) {   // prefetch next Y tile into the other buffer
      const bf16_t* ynext = y_ws + ((size_t)(tt + 1) * R_ + r0) * F_;
      async_copy_b64(ynext + threadIdx.x * 4, &y_l[(buf ^ 1) * 512 + threadIdx.x * 4]);
      asm volatile("s_wait_asynccnt 0x1" ::: "memory");
    } else {
      asm volatile("s_wait_asynccnt 0x0" ::: "memory");
    }
    __syncthreads();     // current Y tile (all waves' portions) visible

    // Y A-fragment (16 rows x K=32) from the staged LDS tile.
    const bf16_t* yp = y_l + buf * 512 + ln * F_ + kh * 8;
    const bf16x16 yf = ldpair(yp, yp + 16);

    // h_{t-1} A-fragments for the 4 K-blocks of the recurrent matmul.
    bf16x16 hf[4];
    #pragma unroll
    for (int kb = 0; kb < 4; ++kb) {
      const bf16_t* hp = h_s + ln * HP + kb * 32 + kh * 8;
      hf[kb] = ldpair(hp, hp + 16);
    }

    // Gates: per (gate q, tile u): 1 WMMA (x path) + 4 WMMAs (h path).
    f32x8 g[4][2];
    #pragma unroll
    for (int q = 0; q < 4; ++q)
      #pragma unroll
      for (int u = 0; u < 2; ++u) {
        const int col = q * H_ + wave * 32 + u * 16 + ln;
        f32x8 acc;
        #pragma unroll
        for (int r = 0; r < 8; ++r) acc[r] = bias[q][u];
        const bf16_t* wp = wih_p + (size_t)col * F_ + kh * 8;
        acc = wmma_bf16(yf, ldpair(wp, wp + 16), acc);
        #pragma unroll
        for (int kb = 0; kb < 4; ++kb) {
          const bf16_t* wp2 = whh_p + (size_t)col * H_ + kb * 32 + kh * 8;
          acc = wmma_bf16(hf[kb], ldpair(wp2, wp2 + 16), acc);
        }
        g[q][u] = acc;
      }

    __syncthreads();  // all reads of h_{t-1} from LDS complete

    // Lane-local LSTM cell update; publish h_t (bf16) to LDS.
    const bool last = (tt == Tc - 1);
    #pragma unroll
    for (int u = 0; u < 2; ++u) {
      const int j = wave * 32 + u * 16 + ln;
      #pragma unroll
      for (int r = 0; r < 8; ++r) {
        const float iv = sigm(g[0][u][r]);
        const float fv = sigm(g[1][u][r]);
        const float gv = tanhf(g[2][u][r]);
        const float ov = sigm(g[3][u][r]);
        const float cv = fv * cfr[u][r] + iv * gv;
        cfr[u][r] = cv;
        const float hv = ov * tanhf(cv);
        const int mr = kh * 8 + r;
        h_s[mr * HP + j] = (bf16_t)hv;
        if (last) {  // persist state for the next chunk
          h_ws[(size_t)(r0 + mr) * H_ + j] = hv;
          c_ws[(size_t)(r0 + mr) * H_ + j] = cv;
        }
      }
    }
    __syncthreads();  // h_t visible to all waves

    // Dense head: waves 0..1 each produce one 16x16 output tile (K=128).
    if (wave < 2) {
      f32x8 acc;
      #pragma unroll
      for (int r = 0; r < 8; ++r) acc[r] = dbias;
      #pragma unroll
      for (int kb = 0; kb < 4; ++kb) {
        const bf16_t* hp = h_s + ln * HP + kb * 32 + kh * 8;
        const bf16_t* dp = dw_p + (size_t)ocol * H_ + kb * 32 + kh * 8;
        acc = wmma_bf16(ldpair(hp, hp + 16), ldpair(dp, dp + 16), acc);
      }
      const int t = chunk * Tc + tt;
      #pragma unroll
      for (int r = 0; r < 8; ++r) {
        const int mr = kh * 8 + r;
        out[(((size_t)bb * T_ + t) * M_ + m0 + mr) * O_ + ocol] = acc[r];
      }
    }
  }
}

// ---------------------------------------------------------------------------
extern "C" void kernel_launch(void* const* d_in, const int* in_sizes, int n_in,
                              void* d_out, int out_size, void* d_ws, size_t ws_size,
                              hipStream_t stream) {
  const float* x       = (const float*)d_in[0];
  const float* assoc_W = (const float*)d_in[1];
  const float* W_ih    = (const float*)d_in[2];
  const float* W_hh    = (const float*)d_in[3];
  const float* b_ih    = (const float*)d_in[4];
  const float* b_hh    = (const float*)d_in[5];
  const float* dense_W = (const float*)d_in[6];
  const float* dense_b = (const float*)d_in[7];
  float* out = (float*)d_out;
  (void)in_sizes; (void)n_in; (void)out_size;

  auto al = [](size_t v) { return (v + 255) & ~(size_t)255; };

  const size_t hsz = (size_t)R_ * H_ * sizeof(float);
  const size_t fixed = al(hsz) + al(hsz) + al((size_t)F_ * M_ * M_ * 2) +
                       al((size_t)G_ * F_ * 2) + al((size_t)G_ * H_ * 2) +
                       al((size_t)O_ * H_ * 2);

  // Largest chunk length (divisor of 168) whose y+u buffers fit the workspace.
  static const int tc_opts[] = {28, 24, 21, 14, 12, 8, 7, 6, 4, 3, 2, 1};
  int Tc = 1;
  for (int i = 0; i < 12; ++i) {
    const size_t per = al((size_t)tc_opts[i] * R_ * F_ * 2);   // y and u each
    if (fixed + 2 * per <= ws_size) { Tc = tc_opts[i]; break; }
  }

  char* wsp = (char*)d_ws;
  bf16_t* y_ws  = (bf16_t*)wsp;  wsp += al((size_t)Tc * R_ * F_ * 2);
  bf16_t* u_ws  = (bf16_t*)wsp;  wsp += al((size_t)Tc * R_ * F_ * 2);
  float*  h_ws  = (float*)wsp;   wsp += al(hsz);
  float*  c_ws  = (float*)wsp;   wsp += al(hsz);
  bf16_t* aW_p  = (bf16_t*)wsp;  wsp += al((size_t)F_ * M_ * M_ * 2);
  bf16_t* wih_p = (bf16_t*)wsp;  wsp += al((size_t)G_ * F_ * 2);
  bf16_t* whh_p = (bf16_t*)wsp;  wsp += al((size_t)G_ * H_ * 2);
  bf16_t* dw_p  = (bf16_t*)wsp;

  pack_k<<<(F_ * M_ * M_ + 255) / 256, 256, 0, stream>>>(
      assoc_W, W_ih, W_hh, dense_W, aW_p, wih_p, whh_p, dw_p);

  const int nchunks = T_ / Tc;
  for (int chunk = 0; chunk < nchunks; ++chunk) {
    squash_k<<<Tc * 64, 256, 0, stream>>>(x, u_ws, chunk, Tc);
    icfa_k<<<128 * Tc, 128, 0, stream>>>(aW_p, u_ws, y_ws, Tc);
    lstm_scan_k<<<R_ / 16, 128, 0, stream>>>(y_ws, wih_p, whh_p, dw_p,
                                             b_ih, b_hh, dense_b, out,
                                             h_ws, c_ws, chunk, Tc);
  }
}
